// Block_9483287789889
// MI455X (gfx1250) — matrix operand order repaired and checked by reading, not compiled, not run on hardware
//
#include <hip/hip_runtime.h>

#define B_  4
#define T_  2048
#define C_  384
#define H_  6
#define HS_ 64
#define F_  1536
#define BT_ (B_*T_)

typedef __attribute__((ext_vector_type(16))) __bf16 v16bf;
typedef __attribute__((ext_vector_type(8)))  float  v8f;
typedef __attribute__((ext_vector_type(4)))  unsigned int v4u;

union Frag {
  v16bf bf;
  v4u   q[2];
};

__device__ __forceinline__ unsigned short f2bf(float f) {
  unsigned int u = __float_as_uint(f);
  u += 0x7FFFu + ((u >> 16) & 1u);   // round-to-nearest-even
  return (unsigned short)(u >> 16);
}

// A-fragment (16x32 bf16): lane<16 -> row=lane, K {0..7,16..23}; lane>=16 -> row=lane-16, K {8..15,24..31}
__device__ __forceinline__ v16bf load_a_frag(const unsigned short* rowPtr, int lane) {
  const int off = (lane < 16) ? 0 : 8;
  Frag f;
  f.q[0] = *(const v4u*)(rowPtr + off);
  f.q[1] = *(const v4u*)(rowPtr + off + 16);
  return f.bf;
}

// B-fragment from Bt[n][k] (row-major over K): lane<16 -> col=lane, K {0..7,16..23}; lane>=16 -> col=lane-16, K {8..15,24..31}
__device__ __forceinline__ v16bf load_b_frag(const unsigned short* rowPtr, int lane) {
  const int off = (lane < 16) ? 0 : 8;
  Frag f;
  f.q[0] = *(const v4u*)(rowPtr + off);
  f.q[1] = *(const v4u*)(rowPtr + off + 16);
  return f.bf;
}

__device__ __forceinline__ v8f wmma_bf16(v16bf a, v16bf b, v8f c) {
  return __builtin_amdgcn_wmma_f32_16x16x32_bf16(false, a, false, b, (short)0, c, false, false);
}

// ---------------- weight convert + transpose: in (R x N) f32 -> out (N x R) bf16, batched over grid.y
__global__ void tconv_kernel(const float* __restrict__ in, unsigned short* __restrict__ out,
                             int R, int N) {
  size_t bofs = (size_t)blockIdx.y * R * N;
  int idx = blockIdx.x * 256 + threadIdx.x;
  if (idx < R * N) {
    int r = idx / N, c = idx % N;
    out[bofs + (size_t)c * R + r] = f2bf(in[bofs + idx]);
  }
}

// ---------------- layernorm: one wave32 per row of C_=384, bf16 output
__global__ void ln_kernel(const float* __restrict__ x, const float* __restrict__ g,
                          const float* __restrict__ be, unsigned short* __restrict__ out) {
  int lane = threadIdx.x & 31;
  int wave = threadIdx.x >> 5;
  int row  = blockIdx.x * (blockDim.x >> 5) + wave;
  const float* xr = x + (size_t)row * C_;
  float vals[C_ / 32];
  float s = 0.f, s2 = 0.f;
#pragma unroll
  for (int i = 0; i < C_ / 32; ++i) {
    float v = xr[lane + i * 32];
    vals[i] = v; s += v; s2 += v * v;
  }
#pragma unroll
  for (int off = 16; off; off >>= 1) {
    s  += __shfl_xor(s,  off);
    s2 += __shfl_xor(s2, off);
  }
  float mean = s  * (1.f / C_);
  float var  = s2 * (1.f / C_) - mean * mean;
  float rstd = rsqrtf(var + 1e-5f);
  unsigned short* orow = out + (size_t)row * C_;
#pragma unroll
  for (int i = 0; i < C_ / 32; ++i) {
    int c = lane + i * 32;
    orow[c] = f2bf((vals[i] - mean) * rstd * g[c] + be[c]);
  }
}

// ---------------- QKV projection GEMM, 4x2 register-blocked (64x32 output per wave).
// grid: (8, B*H, 3), block 256 (8 waves). q,k row-major [bh][t][d] bf16 (q pre-scaled by C^-1/2);
// v stored transposed [bh][d][t] bf16.
__global__ void qkv_kernel(const unsigned short* __restrict__ hb,
                           const unsigned short* __restrict__ wqt,
                           const unsigned short* __restrict__ wkt,
                           const unsigned short* __restrict__ wvt,
                           unsigned short* __restrict__ qout,
                           unsigned short* __restrict__ kout,
                           unsigned short* __restrict__ vtout) {
  int lane = threadIdx.x & 31;
  int wave = threadIdx.x >> 5;
  int bh   = blockIdx.y;
  int b = bh / H_, h = bh % H_;
  int mat  = blockIdx.z;
  int blk  = blockIdx.x * 8 + wave;        // 0..63
  int bm   = blk >> 1;                     // 64-row block of T (32)
  int bn   = blk & 1;                      // 32-col block of HS (2)

  const unsigned short* W = (mat == 0 ? wqt : (mat == 1 ? wkt : wvt)) + (size_t)h * HS_ * C_;
  const unsigned short* arow = hb + ((size_t)b * T_ + bm * 64 + (lane & 15)) * C_;
  const unsigned short* brow = W + ((size_t)bn * 32 + (lane & 15)) * C_;

  v8f acc[4][2] = {};
  for (int k0 = 0; k0 < C_; k0 += 32) {
    v16bf af[4], bfr[2];
#pragma unroll
    for (int i = 0; i < 4; ++i) af[i]  = load_a_frag(arow + (size_t)(16 * i) * C_ + k0, lane);
#pragma unroll
    for (int j = 0; j < 2; ++j) bfr[j] = load_b_frag(brow + (size_t)(16 * j) * C_ + k0, lane);
#pragma unroll
    for (int i = 0; i < 4; ++i)
#pragma unroll
      for (int j = 0; j < 2; ++j)
        acc[i][j] = wmma_bf16(af[i], bfr[j], acc[i][j]);
  }

  float scale = (mat == 0) ? rsqrtf((float)C_) : 1.0f;
  int n16 = lane & 15;
  int mbase = (lane < 16) ? 0 : 8;
  if (mat < 2) {
    unsigned short* base = (mat == 0 ? qout : kout) + (size_t)bh * T_ * HS_;
#pragma unroll
    for (int i = 0; i < 4; ++i)
#pragma unroll
      for (int j = 0; j < 2; ++j) {
        int t0 = bm * 64 + i * 16 + mbase;
        int d  = bn * 32 + j * 16 + n16;
#pragma unroll
        for (int r = 0; r < 8; ++r)
          base[(size_t)(t0 + r) * HS_ + d] = f2bf(acc[i][j][r] * scale);
      }
  } else {
    // transposed store: vt[bh][d][t], 8 contiguous t values -> one 16B store
#pragma unroll
    for (int i = 0; i < 4; ++i)
#pragma unroll
      for (int j = 0; j < 2; ++j) {
        union { unsigned short s[8]; v4u q; } pk;
#pragma unroll
        for (int r = 0; r < 8; ++r) pk.s[r] = f2bf(acc[i][j][r]);
        unsigned short* outp = vtout + ((size_t)bh * HS_ + bn * 32 + j * 16 + n16) * T_ +
                               bm * 64 + i * 16 + mbase;
        *(v4u*)outp = pk.q;
      }
  }
}

// ---------------- pass A: per-key-column softmax stats over the query axis (t >= s).
// grid (16, B*H), block 256: one wave per 16-wide s-tile.
__global__ void stats_kernel(const unsigned short* __restrict__ q,
                             const unsigned short* __restrict__ kk,
                             float* __restrict__ mcol, float* __restrict__ lcol) {
  int lane  = threadIdx.x & 31;
  int wave  = threadIdx.x >> 5;
  int bh    = blockIdx.y;
  int stile = blockIdx.x * 8 + wave;       // 0..127
  int s     = stile * 16 + (lane & 15);

  const unsigned short* qbase = q + (size_t)bh * T_ * HS_;
  const unsigned short* krow  = kk + ((size_t)bh * T_ + s) * HS_;   // Bt row: n=s over k=d
  v16bf kb0 = load_b_frag(krow, lane);
  v16bf kb1 = load_b_frag(krow + 32, lane);

  float runm = -3.0e38f, runl = 0.f;
  int mbase = (lane < 16) ? 0 : 8;
  for (int tt = stile; tt < T_ / 16; ++tt) {
    const unsigned short* qrow = qbase + ((size_t)tt * 16 + (lane & 15)) * HS_;
    v8f acc = {};
    acc = wmma_bf16(load_a_frag(qrow, lane),      kb0, acc);
    acc = wmma_bf16(load_a_frag(qrow + 32, lane), kb1, acc);
#pragma unroll
    for (int r = 0; r < 8; ++r) {
      int t = tt * 16 + mbase + r;
      if (t >= s) {                        // causal mask
        float sc = acc[r];
        if (sc > runm) { runl *= __expf(runm - sc); runm = sc; }
        runl += __expf(sc - runm);
      }
    }
  }
  // merge the two half-columns (lane L and L+16 hold different row halves of column L)
  float om = __shfl_xor(runm, 16);
  float ol = __shfl_xor(runl, 16);
  float nm = fmaxf(runm, om);
  float nl = runl * __expf(runm - nm) + ol * __expf(om - nm);
  if (lane < 16) {
    mcol[(size_t)bh * T_ + s] = nm;
    lcol[(size_t)bh * T_ + s] = nl;
  }
}

// ---------------- pass B: recompute score tiles, normalize per column, P x V via WMMA.
// grid (32, B*H), block 128: one wave per 16-row t-tile; writes attn concat-head bf16 [BT][C].
__global__ void attnout_kernel(const unsigned short* __restrict__ q,
                               const unsigned short* __restrict__ kk,
                               const unsigned short* __restrict__ vt,
                               const float* __restrict__ mcol,
                               const float* __restrict__ lcol,
                               unsigned short* __restrict__ attn) {
  __shared__ __attribute__((aligned(16))) unsigned short pt[4][16 * 40];
  int lane  = threadIdx.x & 31;
  int wave  = threadIdx.x >> 5;            // 0..3
  int bh    = blockIdx.y;
  int b = bh / H_, h = bh % H_;
  int ttile = blockIdx.x * 4 + wave;       // 0..127

  const unsigned short* qrow = q + ((size_t)bh * T_ + ttile * 16 + (lane & 15)) * HS_;
  v16bf qa0 = load_a_frag(qrow, lane);
  v16bf qa1 = load_a_frag(qrow + 32, lane);

  v8f o[4] = {{}, {}, {}, {}};
  int mbase = (lane < 16) ? 0 : 8;
  int nschunks = (ttile * 16 + 15) / 32 + 1;     // s-chunks of 32 covering s <= t_max

  for (int sc = 0; sc < nschunks; ++sc) {
    int s0 = sc * 32;
    // compute & normalize two 16x16 score tiles into the LDS p-tile (t x 32, bf16)
#pragma unroll
    for (int half = 0; half < 2; ++half) {
      int ss = s0 + half * 16 + (lane & 15);
      const unsigned short* krow = kk + ((size_t)bh * T_ + ss) * HS_;
      v8f acc = {};
      acc = wmma_bf16(qa0, load_b_frag(krow, lane), acc);
      acc = wmma_bf16(qa1, load_b_frag(krow + 32, lane), acc);
      float mval = mcol[(size_t)bh * T_ + ss];
      float linv = 1.0f / lcol[(size_t)bh * T_ + ss];
#pragma unroll
      for (int r = 0; r < 8; ++r) {
        int t = ttile * 16 + mbase + r;
        float p = (ss <= t) ? __expf(acc[r] - mval) * linv : 0.f;
        pt[wave][(mbase + r) * 40 + half * 16 + (lane & 15)] = f2bf(p);
      }
    }
    // wave-internal LDS write->read fence (no block barrier: waves have different trip counts)
    __asm__ volatile("s_wait_dscnt 0" ::: "memory");
    // re-load p as an A fragment
    int m = lane & 15;
    int koff = (lane < 16) ? 0 : 8;
    Frag pf;
    pf.q[0] = *(const v4u*)&pt[wave][m * 40 + koff];
    pf.q[1] = *(const v4u*)&pt[wave][m * 40 + 16 + koff];
    // P (16x32) x V (32x64): Bt rows are vt[bh][d][s] (contiguous in s)
#pragma unroll
    for (int nt = 0; nt < 4; ++nt) {
      const unsigned short* vrow = vt + ((size_t)bh * HS_ + nt * 16 + (lane & 15)) * T_ + s0;
      o[nt] = wmma_bf16(pf.bf, load_b_frag(vrow, lane), o[nt]);
    }
  }
  // store concat-head bf16: attn[(b*T + t)][h*HS + d]
  int n = lane & 15;
#pragma unroll
  for (int nt = 0; nt < 4; ++nt)
#pragma unroll
    for (int r = 0; r < 8; ++r)
      attn[((size_t)b * T_ + ttile * 16 + mbase + r) * C_ + h * HS_ + nt * 16 + n] = f2bf(o[nt][r]);
}

// ---------------- generic WMMA GEMM, 4x2 register-blocked (64x32 output per wave),
// with bias / residual / ReLU epilogue. A: M x K bf16 row-major; Bt: N x K bf16 row-major.
__global__ void gemm_kernel(const unsigned short* __restrict__ A,
                            const unsigned short* __restrict__ Bt,
                            const float* __restrict__ bias,
                            const float* __restrict__ res,
                            float* __restrict__ outF,
                            unsigned short* __restrict__ outB,
                            int K, int N, int relu) {
  int lane = threadIdx.x & 31;
  int wave = threadIdx.x >> 5;
  int blk  = blockIdx.x * 8 + wave;
  int nblocks = N >> 5;                // 32-wide N blocks
  int bm = blk / nblocks;
  int bn = blk % nblocks;

  const unsigned short* arow = A  + ((size_t)bm * 64 + (lane & 15)) * K;
  const unsigned short* brow = Bt + ((size_t)bn * 32 + (lane & 15)) * K;

  v8f acc[4][2] = {};
  for (int k0 = 0; k0 < K; k0 += 32) {
    v16bf af[4], bfr[2];
#pragma unroll
    for (int i = 0; i < 4; ++i) af[i]  = load_a_frag(arow + (size_t)(16 * i) * K + k0, lane);
#pragma unroll
    for (int j = 0; j < 2; ++j) bfr[j] = load_b_frag(brow + (size_t)(16 * j) * K + k0, lane);
#pragma unroll
    for (int i = 0; i < 4; ++i)
#pragma unroll
      for (int j = 0; j < 2; ++j)
        acc[i][j] = wmma_bf16(af[i], bfr[j], acc[i][j]);
  }

  int n16 = lane & 15;
  int mbase = (lane < 16) ? 0 : 8;
#pragma unroll
  for (int j = 0; j < 2; ++j) {
    int n = bn * 32 + j * 16 + n16;
    float bv = bias ? bias[n] : 0.f;
#pragma unroll
    for (int i = 0; i < 4; ++i) {
      int m0 = bm * 64 + i * 16 + mbase;
#pragma unroll
      for (int r = 0; r < 8; ++r) {
        size_t idx = (size_t)(m0 + r) * N + n;
        float v = acc[i][j][r] + bv;
        if (res)  v += res[idx];
        if (relu) v = fmaxf(v, 0.f);
        if (outF) outF[idx] = v;
        if (outB) outB[idx] = f2bf(v);
      }
    }
  }
}

extern "C" void kernel_launch(void* const* d_in, const int* in_sizes, int n_in,
                              void* d_out, int out_size, void* d_ws, size_t ws_size,
                              hipStream_t stream) {
  (void)in_sizes; (void)n_in; (void)out_size; (void)ws_size;
  const float* x      = (const float*)d_in[0];
  const float* wq     = (const float*)d_in[1];
  const float* wk     = (const float*)d_in[2];
  const float* wv     = (const float*)d_in[3];
  const float* w_proj = (const float*)d_in[4];
  const float* b_proj = (const float*)d_in[5];
  const float* w1     = (const float*)d_in[6];
  const float* b1     = (const float*)d_in[7];
  const float* w2     = (const float*)d_in[8];
  const float* b2     = (const float*)d_in[9];
  const float* g1     = (const float*)d_in[10];
  const float* be1    = (const float*)d_in[11];
  const float* g2     = (const float*)d_in[12];
  const float* be2    = (const float*)d_in[13];
  float* out = (float*)d_out;

  char* wsp = (char*)d_ws;
  auto carve = [&](size_t bytes) -> void* {
    void* p = (void*)wsp;
    wsp += (bytes + 255) & ~((size_t)255);
    return p;
  };
  unsigned short* hb    = (unsigned short*)carve((size_t)BT_ * C_ * 2);
  unsigned short* h2b   = (unsigned short*)carve((size_t)BT_ * C_ * 2);
  unsigned short* attnb = (unsigned short*)carve((size_t)BT_ * C_ * 2);
  unsigned short* actb  = (unsigned short*)carve((size_t)BT_ * F_ * 2);
  unsigned short* qb    = (unsigned short*)carve((size_t)B_ * H_ * T_ * HS_ * 2);
  unsigned short* kb    = (unsigned short*)carve((size_t)B_ * H_ * T_ * HS_ * 2);
  unsigned short* vtb   = (unsigned short*)carve((size_t)B_ * H_ * T_ * HS_ * 2);
  float*          y1    = (float*)carve((size_t)BT_ * C_ * 4);
  float*          mcol  = (float*)carve((size_t)B_ * H_ * T_ * 4);
  float*          lcol  = (float*)carve((size_t)B_ * H_ * T_ * 4);
  unsigned short* wqt   = (unsigned short*)carve((size_t)H_ * HS_ * C_ * 2);
  unsigned short* wkt   = (unsigned short*)carve((size_t)H_ * HS_ * C_ * 2);
  unsigned short* wvt   = (unsigned short*)carve((size_t)H_ * HS_ * C_ * 2);
  unsigned short* wpt   = (unsigned short*)carve((size_t)C_ * C_ * 2);
  unsigned short* w1t   = (unsigned short*)carve((size_t)C_ * F_ * 2);
  unsigned short* w2t   = (unsigned short*)carve((size_t)F_ * C_ * 2);

  // weight convert + transpose to [N][K] bf16
  tconv_kernel<<<dim3((C_ * HS_ + 255) / 256, H_), 256, 0, stream>>>(wq, wqt, C_, HS_);
  tconv_kernel<<<dim3((C_ * HS_ + 255) / 256, H_), 256, 0, stream>>>(wk, wkt, C_, HS_);
  tconv_kernel<<<dim3((C_ * HS_ + 255) / 256, H_), 256, 0, stream>>>(wv, wvt, C_, HS_);
  tconv_kernel<<<dim3((C_ * C_  + 255) / 256, 1 ), 256, 0, stream>>>(w_proj, wpt, C_, C_);
  tconv_kernel<<<dim3((C_ * F_  + 255) / 256, 1 ), 256, 0, stream>>>(w1, w1t, C_, F_);
  tconv_kernel<<<dim3((F_ * C_  + 255) / 256, 1 ), 256, 0, stream>>>(w2, w2t, F_, C_);

  // LN1
  ln_kernel<<<dim3(BT_ / 8), 256, 0, stream>>>(x, g1, be1, hb);
  // QKV (q pre-scaled by C^-1/2, V stored transposed), 64x32 per wave
  qkv_kernel<<<dim3(8, B_ * H_, 3), 256, 0, stream>>>(hb, wqt, wkt, wvt, qb, kb, vtb);
  // query-axis softmax column stats
  stats_kernel<<<dim3(16, B_ * H_), 256, 0, stream>>>(qb, kb, mcol, lcol);
  // attention output (recompute scores, normalize, P x V)
  attnout_kernel<<<dim3(32, B_ * H_), 128, 0, stream>>>(qb, kb, vtb, mcol, lcol, attnb);
  // out-proj + bias + residual1 -> y1 (f32): (8192/64)*(384/32)/8 = 192 blocks
  gemm_kernel<<<dim3((BT_ / 64) * (C_ / 32) / 8), 256, 0, stream>>>(
      attnb, wpt, b_proj, x, y1, (unsigned short*)nullptr, C_, C_, 0);
  // LN2
  ln_kernel<<<dim3(BT_ / 8), 256, 0, stream>>>(y1, g2, be2, h2b);
  // fc1 + bias + ReLU -> bf16 activations: (8192/64)*(1536/32)/8 = 768 blocks
  gemm_kernel<<<dim3((BT_ / 64) * (F_ / 32) / 8), 256, 0, stream>>>(
      h2b, w1t, b1, (const float*)nullptr, (float*)nullptr, actb, C_, F_, 1);
  // fc2 + bias + residual2 -> final output (f32)
  gemm_kernel<<<dim3((BT_ / 64) * (C_ / 32) / 8), 256, 0, stream>>>(
      actb, w2t, b2, y1, out, (unsigned short*)nullptr, F_, C_, 0);
}
